// Attention_19258633356067
// MI455X (gfx1250) — matrix operand order repaired and checked by reading, not compiled, not run on hardware
//
#include <hip/hip_runtime.h>
#include <hip/hip_bf16.h>

// ---- problem constants (match reference) ----
#define BB    32
#define TK    1024
#define NS    64
#define HID2  1024
#define ENC   2048

typedef __attribute__((ext_vector_type(16))) __bf16 v16bf;
typedef __attribute__((ext_vector_type(2)))  __bf16 bf16x2;
typedef __attribute__((ext_vector_type(8)))  float  v8f;
typedef __attribute__((ext_vector_type(2)))  float  f32x2;

union FragU {
    uint4 q[2];
    v16bf v;
};

__device__ __forceinline__ unsigned pack_bf16(float x, float y) {
    union { bf16x2 h2; unsigned u; } cv;
    cv.h2.x = (__bf16)x;
    cv.h2.y = (__bf16)y;
    return cv.u;
}

// ---------------------------------------------------------------------------
// f32 -> packed bf16 conversion (weights), 2 elements / thread
// ---------------------------------------------------------------------------
__global__ __launch_bounds__(256) void cvt_bf16_kernel(const float* __restrict__ src,
                                                       unsigned* __restrict__ dst, int n_pairs) {
    int i = blockIdx.x * 256 + threadIdx.x;
    if (i < n_pairs) {
        f32x2 f = reinterpret_cast<const f32x2*>(src)[i];
        dst[i] = pack_bf16(f.x, f.y);
    }
}

// ---------------------------------------------------------------------------
// dec_fea[b,n] = s_t_hat[b,:] . W_dec[n,:] + b_dec[n]
// ---------------------------------------------------------------------------
__global__ __launch_bounds__(256) void dec_fea_kernel(const float* __restrict__ sth,
                                                      const float* __restrict__ Wdec,
                                                      const float* __restrict__ bdec,
                                                      float* __restrict__ dec) {
    __shared__ float sv[HID2];
    const int b = blockIdx.x;
    const int n = blockIdx.y * 256 + threadIdx.x;
    for (int i = threadIdx.x; i < HID2; i += 256) sv[i] = sth[(size_t)b * HID2 + i];
    __syncthreads();
    float acc = bdec[n];
    const float* w = Wdec + (size_t)n * HID2;
    for (int k = 0; k < HID2; k += 4) {
        float4 wf = *reinterpret_cast<const float4*>(w + k);
        acc += wf.x * sv[k] + wf.y * sv[k + 1] + wf.z * sv[k + 2] + wf.w * sv[k + 3];
    }
    dec[(size_t)b * HID2 + n] = acc;
}

// ---------------------------------------------------------------------------
// Fused WMMA GEMM -> tanh -> dot(v) score kernel (token path).
// Workgroup: 256 threads = 8 waves. 32 output rows (2 M-tiles) per workgroup.
// Wave w covers N in [128w, 128w+128): 8 N-tiles; B fragment reused for 2 WMMAs.
// All loop-carried addressing is uniform (SALU); per-lane offsets are fixed,
// so no WMMA<->VALU WAR hazard NOPs. A tile staged to LDS as bf16, row stride
// 20 dwords (conflict-free, 16B-aligned b128 fragment loads). NT streaming.
// ---------------------------------------------------------------------------
__global__ __launch_bounds__(256) void token_scores_kernel(
    const float* __restrict__ h,            // [B,TK,ENC]
    const float* __restrict__ s,            // [B,TK,HID2]
    const unsigned short* __restrict__ WhB, // [HID2,ENC]  bf16
    const unsigned short* __restrict__ WsB, // [HID2,HID2] bf16
    const float* __restrict__ dec_fea,      // [B,HID2]
    const float* __restrict__ coverage,     // [B,TK]
    const float* __restrict__ Wc,           // [HID2]
    const float* __restrict__ vvec,         // [HID2]
    float* __restrict__ scores_raw)         // [B,TK]
{
    __shared__ unsigned ldsA[32 * 20];
    __shared__ float red[8 * 32];

    const int b    = blockIdx.x;
    const int t0   = blockIdx.y * 32;
    const int tid  = threadIdx.x;
    const int wave = tid >> 5;
    const int lane = tid & 31;
    const int lrow = lane & 15;
    const int hi   = lane >> 4;           // 0 / 1
    const int nbase = wave * 128;

    v8f acc[16];                           // [mt*8 + j]
    for (int j = 0; j < 16; ++j)
        for (int e = 0; e < 8; ++e) acc[j][e] = 0.0f;

    const int srow = tid >> 4;            // staging row (0..15), +16 for second half
    const int spr  = tid & 15;            // staging K-pair

    for (int phase = 0; phase < 2; ++phase) {
        const char* Ab = (const char*)((phase == 0) ? (h + ((size_t)b * TK + t0) * ENC)
                                                    : (s + ((size_t)b * TK + t0) * HID2));
        const char* Wb = (const char*)((phase == 0) ? WhB : WsB);
        const int K = (phase == 0) ? ENC : HID2;

        // fixed per-lane byte offsets; bases advance uniformly (SALU)
        const unsigned aoff0 = ((unsigned)srow * (unsigned)K + (unsigned)(spr * 2)) * 4u;
        const unsigned aoff1 = ((unsigned)(srow + 16) * (unsigned)K + (unsigned)(spr * 2)) * 4u;
        unsigned woff[8];
        for (int j = 0; j < 8; ++j)
            woff[j] = ((unsigned)(nbase + j * 16 + lrow) * (unsigned)K + (unsigned)(hi * 8)) * 2u;

        const int nchunks = K / 32;
        for (int c = 0; c < nchunks; ++c) {
            __syncthreads();
            {   // stage 32x32 A tile as bf16 into LDS (non-temporal stream)
                f32x2 f0 = __builtin_nontemporal_load(reinterpret_cast<const f32x2*>(Ab + aoff0));
                f32x2 f1 = __builtin_nontemporal_load(reinterpret_cast<const f32x2*>(Ab + aoff1));
                ldsA[srow * 20 + spr]        = pack_bf16(f0.x, f0.y);
                ldsA[(srow + 16) * 20 + spr] = pack_bf16(f1.x, f1.y);
            }
            __syncthreads();

            FragU af0, af1;
            {   // A fragments: rows lrow and lrow+16, 2 x b128 each
                const int p0 = hi * 4;
                af0.q[0] = *reinterpret_cast<const uint4*>(&ldsA[lrow * 20 + p0]);
                af0.q[1] = *reinterpret_cast<const uint4*>(&ldsA[lrow * 20 + 8 + p0]);
                af1.q[0] = *reinterpret_cast<const uint4*>(&ldsA[(lrow + 16) * 20 + p0]);
                af1.q[1] = *reinterpret_cast<const uint4*>(&ldsA[(lrow + 16) * 20 + 8 + p0]);
            }
            for (int j = 0; j < 8; ++j) {
                FragU bf_;
                bf_.q[0] = *reinterpret_cast<const uint4*>(Wb + woff[j]);
                bf_.q[1] = *reinterpret_cast<const uint4*>(Wb + woff[j] + 32);
                acc[j] = __builtin_amdgcn_wmma_f32_16x16x32_bf16(
                    false, af0.v, false, bf_.v, (short)0, acc[j], false, false);
                acc[8 + j] = __builtin_amdgcn_wmma_f32_16x16x32_bf16(
                    false, af1.v, false, bf_.v, (short)0, acc[8 + j], false, false);
            }
            Ab += 32 * 4;   // uniform advance
            Wb += 32 * 2;   // uniform advance
        }
    }

    // epilogue: att = acc + dec + cov*Wc ; score += tanh(att)*v
    float covp[16];
    for (int mt = 0; mt < 2; ++mt)
        for (int r = 0; r < 8; ++r)
            covp[mt * 8 + r] = coverage[(size_t)b * TK + t0 + mt * 16 + r + hi * 8];

    float scr[16];
    for (int r = 0; r < 16; ++r) scr[r] = 0.0f;

    for (int j = 0; j < 8; ++j) {
        const int n = nbase + j * 16 + lrow;
        const float dv = dec_fea[(size_t)b * HID2 + n];
        const float wc = Wc[n];
        const float vv = vvec[n];
        for (int mt = 0; mt < 2; ++mt) {
            for (int r = 0; r < 8; ++r) {
                float att = acc[mt * 8 + j][r] + dv + covp[mt * 8 + r] * wc;
                scr[mt * 8 + r] += tanhf(att) * vv;
            }
        }
    }
    for (int mt = 0; mt < 2; ++mt) {
        for (int r = 0; r < 8; ++r) {
            float x = scr[mt * 8 + r];
            x += __shfl_xor(x, 1, 16);
            x += __shfl_xor(x, 2, 16);
            x += __shfl_xor(x, 4, 16);
            x += __shfl_xor(x, 8, 16);
            if (lrow == 0) red[wave * 32 + mt * 16 + r + hi * 8] = x;
        }
    }
    __syncthreads();
    if (tid < 32) {
        float sum = 0.0f;
        for (int w = 0; w < 8; ++w) sum += red[w * 32 + tid];
        scores_raw[(size_t)b * TK + t0 + tid] = sum;
    }
}

// ---------------------------------------------------------------------------
// Sentence scores: ss[b,s] = tanh((rep+dec)@Ws^T) . v2   (16 rows / WG)
// ---------------------------------------------------------------------------
__global__ __launch_bounds__(256) void sent_scores_kernel(
    const float* __restrict__ rep,          // [B,NS,HID2]
    const unsigned short* __restrict__ WsB, // [HID2,HID2] bf16
    const float* __restrict__ dec_fea,      // [B,HID2]
    const float* __restrict__ v2,           // [HID2]
    float* __restrict__ ss)                 // [B,NS]
{
    __shared__ unsigned ldsA[16 * 20];
    __shared__ float red[8 * 16];

    const int r0   = blockIdx.x * 16;     // global row = b*NS + s
    const int b    = r0 / NS;
    const int tid  = threadIdx.x;
    const int wave = tid >> 5;
    const int lane = tid & 31;
    const int lrow = lane & 15;
    const int hi   = lane >> 4;
    const int nbase = wave * 128;

    v8f acc[8];
    for (int j = 0; j < 8; ++j)
        for (int e = 0; e < 8; ++e) acc[j][e] = 0.0f;

    const int srow = tid >> 4;
    const int spr  = tid & 15;

    const char* Ab = (const char*)(rep + (size_t)r0 * HID2);
    const char* Db = (const char*)(dec_fea + (size_t)b * HID2);
    const char* Wb = (const char*)WsB;
    const unsigned aoff = ((unsigned)srow * HID2 + (unsigned)(spr * 2)) * 4u;
    const unsigned doff = (unsigned)(spr * 2) * 4u;
    unsigned woff[8];
    for (int j = 0; j < 8; ++j)
        woff[j] = ((unsigned)(nbase + j * 16 + lrow) * HID2 + (unsigned)(hi * 8)) * 2u;

    for (int c = 0; c < HID2 / 32; ++c) {
        __syncthreads();
        {
            f32x2 f = *reinterpret_cast<const f32x2*>(Ab + aoff);
            f32x2 d = *reinterpret_cast<const f32x2*>(Db + doff);
            ldsA[srow * 20 + spr] = pack_bf16(f.x + d.x, f.y + d.y);
        }
        __syncthreads();

        FragU af;
        const int p0 = hi * 4;
        af.q[0] = *reinterpret_cast<const uint4*>(&ldsA[lrow * 20 + p0]);
        af.q[1] = *reinterpret_cast<const uint4*>(&ldsA[lrow * 20 + 8 + p0]);

        for (int j = 0; j < 8; ++j) {
            FragU bf_;
            bf_.q[0] = *reinterpret_cast<const uint4*>(Wb + woff[j]);
            bf_.q[1] = *reinterpret_cast<const uint4*>(Wb + woff[j] + 32);
            acc[j] = __builtin_amdgcn_wmma_f32_16x16x32_bf16(
                false, af.v, false, bf_.v, (short)0, acc[j], false, false);
        }
        Ab += 32 * 4;
        Db += 32 * 4;
        Wb += 32 * 2;
    }

    float scr[8];
    for (int r = 0; r < 8; ++r) scr[r] = 0.0f;
    for (int j = 0; j < 8; ++j) {
        const int n = nbase + j * 16 + lrow;
        const float vv = v2[n];
        for (int r = 0; r < 8; ++r) scr[r] += tanhf(acc[j][r]) * vv;
    }
    for (int r = 0; r < 8; ++r) {
        float x = scr[r];
        x += __shfl_xor(x, 1, 16);
        x += __shfl_xor(x, 2, 16);
        x += __shfl_xor(x, 4, 16);
        x += __shfl_xor(x, 8, 16);
        if (lrow == 0) red[wave * 16 + r + hi * 8] = x;
    }
    __syncthreads();
    if (tid < 16) {
        float sum = 0.0f;
        for (int w = 0; w < 8; ++w) sum += red[w * 16 + tid];
        ss[r0 + tid] = sum;
    }
}

// ---------------------------------------------------------------------------
// Per-batch combine: softmaxes, M products, head/child, renorm, coverage.
// One block of 1024 threads per batch (thread == token).
// ---------------------------------------------------------------------------
__device__ __forceinline__ float block_sum(float x, float* red) {
    for (int m = 16; m >= 1; m >>= 1) x += __shfl_xor(x, m, 32);
    const int w = threadIdx.x >> 5, ln = threadIdx.x & 31;
    __syncthreads();
    if (ln == 0) red[w] = x;
    __syncthreads();
    float ssum = 0.0f;
    const int nw = blockDim.x >> 5;
    for (int i = 0; i < nw; ++i) ssum += red[i];
    return ssum;
}
__device__ __forceinline__ float block_max(float x, float* red) {
    for (int m = 16; m >= 1; m >>= 1) x = fmaxf(x, __shfl_xor(x, m, 32));
    const int w = threadIdx.x >> 5, ln = threadIdx.x & 31;
    __syncthreads();
    if (ln == 0) red[w] = x;
    __syncthreads();
    float mx = -3.4e38f;
    const int nw = blockDim.x >> 5;
    for (int i = 0; i < nw; ++i) mx = fmaxf(mx, red[i]);
    return mx;
}

__global__ __launch_bounds__(1024) void combine_kernel(
    const float* __restrict__ scores_raw,   // [B,TK]
    const float* __restrict__ ss,           // [B,NS]
    const float* __restrict__ M,            // [B,NS,TK]
    const float* __restrict__ head,         // [B,NS,NS]
    const float* __restrict__ child,        // [B,NS,NS]
    const float* __restrict__ mask,         // [B,TK]
    const float* __restrict__ coverage,     // [B,TK]
    float* __restrict__ attn_dist,          // [B,TK] (output)
    float* __restrict__ cov_out)            // [B,TK] (output)
{
    __shared__ float sc_sh[TK];
    __shared__ float sent_sh[NS];
    __shared__ float nh_sh[NS];
    __shared__ float nc_sh[NS];
    __shared__ float ss_sh[NS];
    __shared__ float red[32];

    const int b = blockIdx.x;
    const int t = threadIdx.x;
    const float mk = mask[(size_t)b * TK + t];

    // scores = softmax(raw) * mask
    float x = scores_raw[(size_t)b * TK + t];
    float mx = block_max(x, red);
    float ex = __expf(x - mx);
    float se = block_sum(ex, red);
    float scores_t = (ex / se) * mk;
    sc_sh[t] = scores_t;
    __syncthreads();

    // sent_att[s] = sum_t M[b,s,t] * scores[t]   (16 threads per sentence)
    {
        const int s_ = t >> 4, j = t & 15;
        const float* Mr = M + ((size_t)b * NS + s_) * TK + j * 64;
        float part = 0.0f;
        for (int tt = 0; tt < 64; ++tt) part += Mr[tt] * sc_sh[j * 64 + tt];
        part += __shfl_xor(part, 1, 16);
        part += __shfl_xor(part, 2, 16);
        part += __shfl_xor(part, 4, 16);
        part += __shfl_xor(part, 8, 16);
        if (j == 0) sent_sh[s_] = part;
    }
    __syncthreads();

    // new_head / new_child (threads 0..63 hold r)
    if (t < NS) {
        float nh = 0.0f, nc = 0.0f;
        for (int s2 = 0; s2 < NS; ++s2) {
            const float sa = sent_sh[s2];
            nh += sa * head[((size_t)b * NS + s2) * NS + t];
            nc += sa * child[((size_t)b * NS + s2) * NS + t];
        }
        nh_sh[t] = nh;
        nc_sh[t] = nc;
        ss_sh[t] = ss[(size_t)b * NS + t];
    }
    __syncthreads();

    // token-level projections through M (one pass over s)
    float tl = 0.0f, ht = 0.0f, ct = 0.0f;
    for (int s2 = 0; s2 < NS; ++s2) {
        const float mm = M[((size_t)b * NS + s2) * TK + t];
        tl += mm * ss_sh[s2];
        ht += mm * nh_sh[s2];
        ct += mm * nc_sh[s2];
    }

    // three softmaxes
    float m1 = block_max(tl, red);  float e1 = __expf(tl - m1); float s1 = block_sum(e1, red);
    float m2 = block_max(ht, red);  float e2 = __expf(ht - m2); float s2s = block_sum(e2, red);
    float m3 = block_max(ct, red);  float e3 = __expf(ct - m3); float s3 = block_sum(e3, red);

    float attn = scores_t * ((e1 / s1) * mk);
    attn += (e2 / s2s) * mk;
    attn += (e3 / s3) * mk;

    float tot = block_sum(attn, red);
    float ad = attn / tot;
    attn_dist[(size_t)b * TK + t] = ad;
    cov_out[(size_t)b * TK + t] = coverage[(size_t)b * TK + t] + ad;
}

// ---------------------------------------------------------------------------
// c_t[b,n] = sum_t attn_dist[b,t] * h[b,t,n]    (bandwidth-bound, NT reads)
// ---------------------------------------------------------------------------
__global__ __launch_bounds__(256) void ctx_kernel(const float* __restrict__ h,
                                                  const float* __restrict__ ad,
                                                  float* __restrict__ ct) {
    __shared__ float a_sh[256];
    const int b = blockIdx.y;
    const int n = blockIdx.x * 256 + threadIdx.x;
    float acc = 0.0f;
    for (int t0 = 0; t0 < TK; t0 += 256) {
        __syncthreads();
        a_sh[threadIdx.x] = ad[(size_t)b * TK + t0 + threadIdx.x];
        __syncthreads();
        const float* hp = h + ((size_t)b * TK + t0) * ENC + n;
        for (int tt = 0; tt < 256; ++tt)
            acc += a_sh[tt] * __builtin_nontemporal_load(hp + (size_t)tt * ENC);
    }
    ct[(size_t)b * ENC + n] = acc;
}

// ---------------------------------------------------------------------------
extern "C" void kernel_launch(void* const* d_in, const int* in_sizes, int n_in,
                              void* d_out, int out_size, void* d_ws, size_t ws_size,
                              hipStream_t stream) {
    (void)in_sizes; (void)n_in; (void)out_size; (void)ws_size;

    const float* s_t_hat  = (const float*)d_in[0];
    const float* h        = (const float*)d_in[1];
    const float* mask     = (const float*)d_in[2];
    const float* coverage = (const float*)d_in[3];
    const float* s        = (const float*)d_in[6];
    const float* M        = (const float*)d_in[7];
    const float* head     = (const float*)d_in[8];
    const float* child    = (const float*)d_in[9];
    const float* rep      = (const float*)d_in[10];
    const float* W_h      = (const float*)d_in[11];
    const float* W_s      = (const float*)d_in[12];
    const float* W_c      = (const float*)d_in[13];
    const float* W_dec    = (const float*)d_in[14];
    const float* b_dec    = (const float*)d_in[15];
    const float* v        = (const float*)d_in[16];
    const float* v2       = (const float*)d_in[17];

    float* out_ct  = (float*)d_out;                 // [B,ENC]
    float* out_ad  = out_ct + (size_t)BB * ENC;     // [B,TK]
    float* out_cov = out_ad + (size_t)BB * TK;      // [B,TK]

    char* ws = (char*)d_ws;
    float* dec_fea    = (float*)ws;  ws += (size_t)BB * HID2 * sizeof(float);
    float* scores_raw = (float*)ws;  ws += (size_t)BB * TK * sizeof(float);
    float* ss         = (float*)ws;  ws += (size_t)BB * NS * sizeof(float);
    unsigned short* WhB = (unsigned short*)ws; ws += (size_t)HID2 * ENC * sizeof(unsigned short);
    unsigned short* WsB = (unsigned short*)ws; ws += (size_t)HID2 * HID2 * sizeof(unsigned short);

    // 1) weights -> bf16 (L2-resident, b128-friendly fragments)
    {
        int np = HID2 * ENC / 2;
        cvt_bf16_kernel<<<(np + 255) / 256, 256, 0, stream>>>(W_h, (unsigned*)WhB, np);
        np = HID2 * HID2 / 2;
        cvt_bf16_kernel<<<(np + 255) / 256, 256, 0, stream>>>(W_s, (unsigned*)WsB, np);
    }
    // 2) dec_fea
    dec_fea_kernel<<<dim3(BB, HID2 / 256), 256, 0, stream>>>(s_t_hat, W_dec, b_dec, dec_fea);
    // 3) sentence scores (WMMA)
    sent_scores_kernel<<<(BB * NS) / 16, 256, 0, stream>>>(rep, WsB, dec_fea, v2, ss);
    // 4) token scores (big fused WMMA kernel, TM=32)
    token_scores_kernel<<<dim3(BB, TK / 32), 256, 0, stream>>>(
        h, s, WhB, WsB, dec_fea, coverage, W_c, v, scores_raw);
    // 5) softmax / M-product combine -> attn_dist, coverage_out
    combine_kernel<<<BB, 1024, 0, stream>>>(scores_raw, ss, M, head, child, mask, coverage,
                                            out_ad, out_cov);
    // 6) context vector
    ctx_kernel<<<dim3(ENC / 256, BB), 256, 0, stream>>>(h, out_ad, out_ct);
}
